// WindowAttention_85864986182048
// MI455X (gfx1250) — matrix operand order repaired and checked by reading, not compiled, not run on hardware
//
#include <hip/hip_runtime.h>
#include <hip/hip_bf16.h>

// ---------------------------------------------------------------------------
// Swin window attention for MI455X (gfx1250, wave32, WMMA bf16 path).
// All WMMA A-operands are stored pre-swizzled in fragment order so each
// A-frag / B-frag is a single contiguous 32B load (no VALU shuffling).
// ---------------------------------------------------------------------------

typedef __bf16 bf16_t;
typedef __attribute__((ext_vector_type(16))) __bf16 v16bf;
typedef __attribute__((ext_vector_type(8)))  __bf16 v8bf;
typedef __attribute__((ext_vector_type(8)))  float  v8f;

#define DIM      256
#define HEADS    8
#define HDIM     32
#define BATCH    32
#define HW       56
#define TOKPAD   64
#define ATTN_SCALE 0.17677669529663687f

__device__ __forceinline__ bf16_t f2bf(float f) { return (bf16_t)f; }

// position of element k inside its 32-wide block, in WMMA A-fragment order:
// contiguous: lane group g = bit4(k), element e = k&15  ->  off = k
__device__ __forceinline__ int aswz(int k) {
  return k;
}

#define WMMA_BF16(A, B, C) \
  __builtin_amdgcn_wmma_f32_16x16x32_bf16(false, (A), false, (B), (short)0, (C), false, false)

// ---------------------------------------------------------------------------
// Kernel 0a: x (fp32, [100352][256]) -> bf16 pre-swizzled A-layout.
// ---------------------------------------------------------------------------
__global__ __launch_bounds__(256)
void k_prep_x(const float* __restrict__ x, bf16_t* __restrict__ xswz)
{
  int t = blockIdx.x * blockDim.x + threadIdx.x;     // 100352*256 threads
  int m = t >> 8, k = t & 255;
  xswz[(size_t)m * DIM + aswz(k)] = f2bf(x[t]);
}

// ---------------------------------------------------------------------------
// Kernel 0b: weights -> bf16 transposed [N][K] (B-frag = contiguous 16).
// ---------------------------------------------------------------------------
__global__ __launch_bounds__(256)
void k_prep_weights(const float* __restrict__ wqkv, const float* __restrict__ wout,
                    bf16_t* __restrict__ wqkvT, bf16_t* __restrict__ woutT)
{
  int t = blockIdx.x * blockDim.x + threadIdx.x;
  if (t < 768 * 256) {
    int n = t >> 8, k = t & 255;
    wqkvT[t] = f2bf(wqkv[(size_t)k * 768 + n]);
  } else {
    int u = t - 768 * 256;
    if (u < 256 * 256) {
      int n = u >> 8, k = u & 255;
      woutT[u] = f2bf(wout[(size_t)k * 256 + n]);
    }
  }
}

// ---------------------------------------------------------------------------
// Kernel 1: zero token-padding (tok 49..63) of Q/K/V.
// ---------------------------------------------------------------------------
__global__ __launch_bounds__(256)
void k_zero_pad(bf16_t* __restrict__ qw, bf16_t* __restrict__ kw, bf16_t* __restrict__ vw)
{
  int t = blockIdx.x * blockDim.x + threadIdx.x;     // 16384*480 threads exactly
  const int per = 15 * 32;
  int id = t / per, slot = t - id * per;
  int tok = 49 + (slot >> 5), d = slot & 31;
  size_t base = (size_t)id * (TOKPAD * HDIM);
  bf16_t z = f2bf(0.0f);
  qw[base + tok * 32 + d] = z;
  kw[base + tok * 32 + d] = z;
  vw[base + d * 64 + tok] = z;
}

// ---------------------------------------------------------------------------
// Kernel 2: QKV GEMM.  [100352,768] = xswz @ wqkvT^T.
// Wave -> 16(M) x 64(N); K-loop: 1 A load + 4 B loads + 4 WMMA, no VALU.
// Epilogue scatters: Q swizzled [id][tok][swz(d)], K [id][tok][d], V [id][d][tok].
// ---------------------------------------------------------------------------
__global__ __launch_bounds__(256)
void k_qkv_gemm(const bf16_t* __restrict__ xswz, const bf16_t* __restrict__ wqkvT,
                bf16_t* __restrict__ qw, bf16_t* __restrict__ kw, bf16_t* __restrict__ vw)
{
  const int lane = threadIdx.x & 31;
  const int wid  = blockIdx.x * 8 + (threadIdx.x >> 5);
  const int tileM = wid / 12;
  const int tileN = wid % 12;
  const int l15 = lane & 15, g = lane >> 4;

  const bf16_t* xrow = xswz + (size_t)(tileM * 16 + l15) * DIM;

  v8f acc[4];
  { v8f z = {}; acc[0] = z; acc[1] = z; acc[2] = z; acc[3] = z; }

  #pragma unroll
  for (int kk = 0; kk < DIM; kk += 32) {
    v16bf a = *(const v16bf*)(xrow + kk + 16 * g);
    const int nb = tileN * 64 + l15;
    #pragma unroll
    for (int nt = 0; nt < 4; ++nt) {
      v16bf b = *(const v16bf*)(wqkvT + (size_t)(nb + nt * 16) * DIM + kk + 16 * g);
      acc[nt] = WMMA_BF16(a, b, acc[nt]);
    }
  }

  #pragma unroll
  for (int nt = 0; nt < 4; ++nt) {
    const int n = tileN * 64 + nt * 16 + l15;
    const int which = n >> 8;            // 0=q 1=k 2=v
    const int hd = (n >> 5) & 7;
    const int d  = n & 31;
    const int dq = aswz(d);              // swizzled position for Q (A operand later)
    v8bf cv = __builtin_convertvector(acc[nt], v8bf);
    #pragma unroll
    for (int r = 0; r < 8; ++r) {
      const int m   = tileM * 16 + r + 8 * g;
      const int bb  = m / 3136;
      const int rem = m - bb * 3136;
      const int yy  = rem / 56, xx = rem - (rem / 56) * 56;
      const int win = (yy / 7) * 8 + (xx / 7);
      const int tok = (yy % 7) * 7 + (xx % 7);
      const size_t id = (size_t)((bb * 8 + hd) * 64 + win);
      if (which == 0)      qw[id * (TOKPAD * HDIM) + tok * 32 + dq] = cv[r];
      else if (which == 1) kw[id * (TOKPAD * HDIM) + tok * 32 + d]  = cv[r];
      else                 vw[id * (TOKPAD * HDIM) + d * 64 + tok]  = cv[r];
    }
  }
}

// ---------------------------------------------------------------------------
// Kernel 3: fused window attention, one wave per (batch, head, window).
// S = Q.K^T (WMMA) -> bias+mask+softmax (shfl_xor) -> P via per-wave LDS
// (stored swizzled => A-frag is one 32B ds load) -> O = P.V (WMMA) ->
// scatter to aout in pre-swizzled A-layout for the final GEMM.
// ---------------------------------------------------------------------------
__global__ __launch_bounds__(256)
void k_window_attn(const bf16_t* __restrict__ qw, const bf16_t* __restrict__ kw,
                   const bf16_t* __restrict__ vw, const float* __restrict__ pos_emb,
                   bf16_t* __restrict__ aswzout)
{
  __shared__ float  pe[169];
  __shared__ bf16_t plds[8][16 * 64];

  for (int t = threadIdx.x; t < 169; t += blockDim.x) pe[t] = pos_emb[t];
  __syncthreads();

  const int lane = threadIdx.x & 31;
  const int w    = threadIdx.x >> 5;
  const int id   = blockIdx.x * 8 + w;
  const int bb   = id >> 9;
  const int hd   = (id >> 6) & 7;
  const int win  = id & 63;
  const int winy = win >> 3, winx = win & 7;
  const int l15 = lane & 15, g = lane >> 4;

  const bf16_t* qb = qw + (size_t)id * (TOKPAD * HDIM);
  const bf16_t* kb = kw + (size_t)id * (TOKPAD * HDIM);
  const bf16_t* vb = vw + (size_t)id * (TOKPAD * HDIM);

  v16bf kfr[4];
  #pragma unroll
  for (int nt = 0; nt < 4; ++nt)
    kfr[nt] = *(const v16bf*)(kb + (nt * 16 + l15) * 32 + 16 * g);
  v16bf vfr[2][2];
  #pragma unroll
  for (int nt = 0; nt < 2; ++nt)
    #pragma unroll
    for (int kt = 0; kt < 2; ++kt)
      vfr[nt][kt] = *(const v16bf*)(vb + (nt * 16 + l15) * 64 + kt * 32 + 16 * g);

  #pragma unroll 1
  for (int mt = 0; mt < 4; ++mt) {
    // Q A-frag: single 32B load (stored pre-swizzled)
    v16bf qa = *(const v16bf*)(qb + (mt * 16 + l15) * 32 + 16 * g);

    v8f s[4];
    #pragma unroll
    for (int nt = 0; nt < 4; ++nt) {
      v8f z = {};
      s[nt] = WMMA_BF16(qa, kfr[nt], z);
    }

    // bias + mask + row softmax (row = 16 lanes of a half-wave)
    #pragma unroll
    for (int r = 0; r < 8; ++r) {
      const int i  = mt * 16 + r + 8 * g;
      const int xi = i / 7, yi = i - xi * 7;
      float vals[4];
      float mx = -3.0e30f;
      #pragma unroll
      for (int nt = 0; nt < 4; ++nt) {
        const int j  = nt * 16 + l15;
        const int xj = j / 7, yj = j - xj * 7;
        float vv = -1.0e30f;
        if (i < 49 && j < 49)
          vv = s[nt][r] * ATTN_SCALE + pe[(xj - xi + 6) * 13 + (yj - yi + 6)];
        vals[nt] = vv;
        mx = fmaxf(mx, vv);
      }
      #pragma unroll
      for (int off = 1; off < 16; off <<= 1)
        mx = fmaxf(mx, __shfl_xor(mx, off, 32));
      float sum = 0.0f;
      #pragma unroll
      for (int nt = 0; nt < 4; ++nt) {
        float p = __expf(vals[nt] - mx);
        vals[nt] = p;
        sum += p;
      }
      #pragma unroll
      for (int off = 1; off < 16; off <<= 1)
        sum += __shfl_xor(sum, off, 32);
      const float rinv = 1.0f / sum;
      #pragma unroll
      for (int nt = 0; nt < 4; ++nt) {
        const int j = nt * 16 + l15;
        plds[w][(r + 8 * g) * 64 + aswz(j)] = f2bf(vals[nt] * rinv);
      }
    }

    // P A-frags: single contiguous 32B LDS loads (DS in-order per wave)
    v16bf pa[2];
    #pragma unroll
    for (int kt = 0; kt < 2; ++kt)
      pa[kt] = *(const v16bf*)(&plds[w][l15 * 64 + kt * 32 + 16 * g]);

    v8f o[2];
    #pragma unroll
    for (int nt = 0; nt < 2; ++nt) {
      v8f z = {};
      z = WMMA_BF16(pa[0], vfr[nt][0], z);
      z = WMMA_BF16(pa[1], vfr[nt][1], z);
      o[nt] = z;
    }

    // scatter to pre-swizzled A layout for the output GEMM
    #pragma unroll
    for (int nt = 0; nt < 2; ++nt) {
      const int c   = hd * 32 + nt * 16 + l15;       // channel 0..255
      const int cof = aswz(c);
      v8bf cv = __builtin_convertvector(o[nt], v8bf);
      #pragma unroll
      for (int r = 0; r < 8; ++r) {
        const int i = mt * 16 + r + 8 * g;
        if (i < 49) {
          const int yy = winy * 7 + i / 7;
          const int xx = winx * 7 + (i % 7);
          const size_t m = (size_t)(bb * 56 + yy) * 56 + xx;
          aswzout[m * DIM + cof] = cv[r];
        }
      }
    }
  }
}

// ---------------------------------------------------------------------------
// Kernel 4: output GEMM + bias.  out[100352,256] = aout @ w_out + b_out (fp32)
// ---------------------------------------------------------------------------
__global__ __launch_bounds__(256)
void k_out_gemm(const bf16_t* __restrict__ aswzout, const bf16_t* __restrict__ woutT,
                const float* __restrict__ b_out, float* __restrict__ out)
{
  const int lane = threadIdx.x & 31;
  const int wid  = blockIdx.x * 8 + (threadIdx.x >> 5);
  const int tileM = wid >> 2;
  const int tileN = wid & 3;
  const int l15 = lane & 15, g = lane >> 4;

  const bf16_t* arow = aswzout + (size_t)(tileM * 16 + l15) * DIM;

  v8f acc[4];
  { v8f z = {}; acc[0] = z; acc[1] = z; acc[2] = z; acc[3] = z; }

  #pragma unroll
  for (int kk = 0; kk < DIM; kk += 32) {
    v16bf a = *(const v16bf*)(arow + kk + 16 * g);
    const int nb = tileN * 64 + l15;
    #pragma unroll
    for (int nt = 0; nt < 4; ++nt) {
      v16bf b = *(const v16bf*)(woutT + (size_t)(nb + nt * 16) * DIM + kk + 16 * g);
      acc[nt] = WMMA_BF16(a, b, acc[nt]);
    }
  }

  #pragma unroll
  for (int nt = 0; nt < 4; ++nt) {
    const int n = tileN * 64 + nt * 16 + l15;
    const float bias = b_out[n];
    #pragma unroll
    for (int r = 0; r < 8; ++r) {
      const int m = tileM * 16 + r + 8 * g;
      out[(size_t)m * DIM + n] = acc[nt][r] + bias;
    }
  }
}

// ---------------------------------------------------------------------------
extern "C" void kernel_launch(void* const* d_in, const int* in_sizes, int n_in,
                              void* d_out, int out_size, void* d_ws, size_t ws_size,
                              hipStream_t stream)
{
  const float* x       = (const float*)d_in[0];   // (32,56,56,256)
  const float* w_qkv   = (const float*)d_in[1];   // (256,768)
  const float* w_out   = (const float*)d_in[2];   // (256,256)
  const float* b_out   = (const float*)d_in[3];   // (256,)
  const float* pos_emb = (const float*)d_in[4];   // (13,13)
  float* out = (float*)d_out;

  char* ws = (char*)d_ws;
  bf16_t* xswz  = (bf16_t*)(ws);                       //  49 MiB [100352][256] swizzled
  bf16_t* qw    = (bf16_t*)(ws + (size_t)51380224);    //  64 MiB [16384][64][32] swizzled-d
  bf16_t* kw    = (bf16_t*)(ws + (size_t)118489088);   //  64 MiB [16384][64][32]
  bf16_t* vw    = (bf16_t*)(ws + (size_t)185597952);   //  64 MiB [16384][32][64]
  bf16_t* aswzo = (bf16_t*)(ws + (size_t)252706816);   //  49 MiB [100352][256] swizzled
  bf16_t* wqkvT = (bf16_t*)(ws + (size_t)304087040);   // 384 KiB [768][256]
  bf16_t* woutT = (bf16_t*)(ws + (size_t)304480256);   // 128 KiB [256][256]

  k_prep_x     <<<100352, 256, 0, stream>>>(x, xswz);
  k_prep_weights<<<1024, 256, 0, stream>>>(w_qkv, w_out, wqkvT, woutT);
  k_zero_pad   <<<30720, 256, 0, stream>>>(qw, kw, vw);
  k_qkv_gemm   <<<9408, 256, 0, stream>>>(xswz, wqkvT, qw, kw, vw);
  k_window_attn<<<2048, 256, 0, stream>>>(qw, kw, vw, pos_emb, aswzo);
  k_out_gemm   <<<3136, 256, 0, stream>>>(aswzo, woutT, b_out, out);
}
